// DirichletEncoder_5420248727595
// MI455X (gfx1250) — matrix operand order repaired and checked by reading, not compiled, not run on hardware
//
#include <hip/hip_runtime.h>
#include <hip/hip_bf16.h>

// ---------------------------------------------------------------------------
// Types for CDNA5 WMMA
// ---------------------------------------------------------------------------
typedef __attribute__((ext_vector_type(16))) __bf16   v16bf;
typedef __attribute__((ext_vector_type(8)))  float    v8f;
typedef __attribute__((ext_vector_type(8)))  unsigned v8u;

// Async global->LDS path (guarded: falls back to plain copies if the
// builtins are absent on this toolchain).
// Probe-derived signature: (int4 AS1*, int4 AS3*, imm offset, imm cpol)
#if defined(__gfx1250__) && __has_builtin(__builtin_amdgcn_global_load_async_to_lds_b128) && __has_builtin(__builtin_amdgcn_s_wait_asynccnt)
#define USE_ASYNC_LDS 1
typedef int v4i_vec __attribute__((vector_size(4 * sizeof(int))));
typedef __attribute__((address_space(1))) v4i_vec GV4;
typedef __attribute__((address_space(3))) v4i_vec LV4;
#else
#define USE_ASYNC_LDS 0
#endif

static __device__ __forceinline__ unsigned short f2bf_rne(float f) {
    unsigned u  = __float_as_uint(f);
    unsigned rb = ((u >> 16) & 1u) + 0x7FFFu;   // round-to-nearest-even
    return (unsigned short)((u + rb) >> 16);
}
static __device__ __forceinline__ unsigned pack2bf(float a, float b) {
    return (unsigned)f2bf_rne(a) | ((unsigned)f2bf_rne(b) << 16);
}

// ---------------------------------------------------------------------------
// Weight pre-pack: fp32 [K][N] -> bf16 K-pair packed, TRANSPOSED: Wp[n][K/2].
//   Wp[n*(K/2)+p] = { bf16(W[2p][n]), bf16(W[2p+1][n]) }
// A lane's 8 B-fragment dwords (pairs 4h..4h+3, 8+4h..8+4h+3) form two b128 runs both in
// global memory and in LDS -> staged with one b128 copy / ds_load_b128 pairs.
// ---------------------------------------------------------------------------
__global__ void pack_weights_t(const float* __restrict__ W, unsigned* __restrict__ Wp,
                               int K, int N) {
    int idx = blockIdx.x * blockDim.x + threadIdx.x;
    int kp  = K / 2;
    int total = N * kp;
    if (idx >= total) return;
    int n = idx / kp, p = idx - n * kp;
    Wp[idx] = pack2bf(W[(size_t)(2 * p) * N + n], W[(size_t)(2 * p + 1) * N + n]);
}

// ---------------------------------------------------------------------------
// Fused GEMM + epilogue, bf16 WMMA (v_wmma_f32_16x16x32_bf16).
//   Block tile: 128(M) x 64(N), K-step 32.  256 threads = 8 waves.
//   Double-buffered LDS; next tile staged via async global->LDS before the
//   current tile's 4 WMMAs -> loads hide behind matrix math, one barrier
//   per K-step.  All fragment ds_loads are issued ahead of the WMMA chain
//   so the DS counter waits can be partial.
//
//   LDS layouts are fragment-major:
//     sA[r][slot], slot order [pairs 0-3 | 8-11 | 4-7 | 12-15] so lane-half hi
//       reads its 8 A-dwords from 2 contiguous b128 loads.
//     sB[n][pair]  (transposed weights) so each lane reads two 4-dword runs: pairs 4hi..+3, 8+4hi..+3.
// ---------------------------------------------------------------------------
#define BM 128
#define BN 64
#define BK 32
#define SAPITCH 20   // 16 used + 4 pad dwords (80B rows keep 16B alignment)
#define SBPITCH 20

template <bool A_FP32, bool FINAL>
__global__ __launch_bounds__(256) void gemm_wmma(
    const void* __restrict__ Asrc, const unsigned* __restrict__ Bpack,
    const float* __restrict__ bias, void* __restrict__ Out,
    int M, int N, int K)
{
    __shared__ unsigned sA[2][BM][SAPITCH];
    __shared__ unsigned sB[2][BN][SBPITCH];

    const int tid      = threadIdx.x;
    const int wave     = tid >> 5;
    const int lane     = tid & 31;
    const int laneN    = lane & 15;
    const int hi       = lane >> 4;              // 0: K 0..7,16..23; 1: K 8..15,24..31
    const int rowBlock = blockIdx.x * BM;
    const int colBlock = blockIdx.y * BN;
    const int waveRow  = wave * 16;

    const int r        = tid >> 1;    // A staging row 0..127
    const int halfsel  = tid & 1;     // which 8-pair half of the K-step
    const int bn_local = tid >> 2;    // B staging column 0..63
    const int bq       = tid & 3;     // B staging pair-quad 0..3

    const int kp = K / 2;             // K in bf16 pairs

    auto stageA = [&](int buf, int kk) {
        if (A_FP32) {
            const float* A = (const float*)Asrc;
            const float4* src =
                (const float4*)(A + (size_t)(rowBlock + r) * K + kk + halfsel * 16);
            float4 f0 = src[0], f1 = src[1], f2 = src[2], f3 = src[3];
            uint4 p0 = { pack2bf(f0.x, f0.y), pack2bf(f0.z, f0.w),
                         pack2bf(f1.x, f1.y), pack2bf(f1.z, f1.w) };
            uint4 p1 = { pack2bf(f2.x, f2.y), pack2bf(f2.z, f2.w),
                         pack2bf(f3.x, f3.y), pack2bf(f3.z, f3.w) };
            // pair p -> slot: [0-3]->0-3, [4-7]->8-11, [8-11]->4-7, [12-15]->12-15
            *(uint4*)&sA[buf][r][halfsel * 4]     = p0;   // pairs h*8+0..3
            *(uint4*)&sA[buf][r][8 + halfsel * 4] = p1;   // pairs h*8+4..7
        } else {
            const unsigned* src =
                (const unsigned*)Asrc + (size_t)(rowBlock + r) * kp + (kk / 2) + halfsel * 8;
#if USE_ASYNC_LDS
            __builtin_amdgcn_global_load_async_to_lds_b128(
                (GV4*)src, (LV4*)&sA[buf][r][halfsel * 4], 0, 0);
            __builtin_amdgcn_global_load_async_to_lds_b128(
                (GV4*)(src + 4), (LV4*)&sA[buf][r][8 + halfsel * 4], 0, 0);
#else
            const uint4* s4 = (const uint4*)src;
            *(uint4*)&sA[buf][r][halfsel * 4]     = s4[0];
            *(uint4*)&sA[buf][r][8 + halfsel * 4] = s4[1];
#endif
        }
    };
    auto stageB = [&](int buf, int kk) {
        const unsigned* src =
            Bpack + (size_t)(colBlock + bn_local) * kp + (kk / 2) + bq * 4;
#if USE_ASYNC_LDS
        __builtin_amdgcn_global_load_async_to_lds_b128(
            (GV4*)src, (LV4*)&sB[buf][bn_local][bq * 4], 0, 0);
#else
        *(uint4*)&sB[buf][bn_local][bq * 4] = *(const uint4*)src;
#endif
    };

    v8f acc[4];
#pragma unroll
    for (int t = 0; t < 4; ++t)
#pragma unroll
        for (int i = 0; i < 8; ++i) acc[t][i] = 0.0f;

    stageA(0, 0);
    stageB(0, 0);
#if USE_ASYNC_LDS
    __builtin_amdgcn_s_wait_asynccnt(0);
#endif
    __syncthreads();

    int buf = 0;
    for (int kk = 0; kk < K; kk += BK) {
        const int nxt = kk + BK;
        if (nxt < K) {                 // stage next tile behind this tile's WMMAs
            stageA(buf ^ 1, nxt);
            stageB(buf ^ 1, nxt);
        }
        if (A_FP32 && kk + 2 * BK < K)  // pull the tile after that toward L2
            __builtin_prefetch((const float*)Asrc + (size_t)(rowBlock + r) * K + kk + 2 * BK, 0, 1);

        // ---- issue ALL fragment loads up front (A + 4 B tiles) ----
        const int m = waveRow + laneN;
        uint4 a0 = *(const uint4*)&sA[buf][m][hi * 8];
        uint4 a1 = *(const uint4*)&sA[buf][m][hi * 8 + 4];
        v8u areg;
        areg[0] = a0.x; areg[1] = a0.y; areg[2] = a0.z; areg[3] = a0.w;
        areg[4] = a1.x; areg[5] = a1.y; areg[6] = a1.z; areg[7] = a1.w;

        v8u breg[4];
#pragma unroll
        for (int t = 0; t < 4; ++t) {
            const int n = t * 16 + laneN;
            uint4 b0 = *(const uint4*)&sB[buf][n][hi * 4];
            uint4 b1 = *(const uint4*)&sB[buf][n][8 + hi * 4];
            breg[t][0] = b0.x; breg[t][1] = b0.y; breg[t][2] = b0.z; breg[t][3] = b0.w;
            breg[t][4] = b1.x; breg[t][5] = b1.y; breg[t][6] = b1.z; breg[t][7] = b1.w;
        }

        // ---- 4 x WMMA back-to-back (partial dscnt waits possible) ----
#pragma unroll
        for (int t = 0; t < 4; ++t) {
            acc[t] = __builtin_amdgcn_wmma_f32_16x16x32_bf16(
                /*neg_a=*/false, __builtin_bit_cast(v16bf, areg),
                /*neg_b=*/false, __builtin_bit_cast(v16bf, breg[t]),
                /*c_mod=*/(short)0, acc[t],
                /*reuse_a=*/false, /*reuse_b=*/false);
        }

        if (nxt < K) {
#if USE_ASYNC_LDS
            __builtin_amdgcn_s_wait_asynccnt(0);
#endif
            __syncthreads();
            buf ^= 1;
        }
    }

    // ---- epilogue (C/D layout: VGPR rr -> M=rr + 8*hi, lane -> N) ----
#pragma unroll
    for (int t = 0; t < 4; ++t) {
        const int col = colBlock + t * 16 + laneN;
        const float bv = bias[col];
#pragma unroll
        for (int rr = 0; rr < 8; ++rr) {
            const int row = rowBlock + waveRow + rr + hi * 8;
            float val = acc[t][rr] + bv;
            if (!FINAL) {
                val = fmaxf(val, 0.0f);
                // pack column pairs across adjacent lanes -> 32-bit stores
                float other = __shfl_xor(val, 1, 32);
                if ((lane & 1) == 0) {
                    unsigned pk = pack2bf(val, other);
                    *(unsigned*)((unsigned short*)Out + (size_t)row * N + col) = pk;
                }
            } else {
                float sp = (val > 20.0f) ? val : log1pf(__expf(val));
                ((float*)Out)[(size_t)row * N + col] = sp + 0.5f;
            }
        }
    }
}

// ---------------------------------------------------------------------------
// Marsaglia-Tsang gamma(alpha,1) sampler: 16 proposals, first accepted wins.
// Deterministic counter-based RNG (splitmix64 of the element index).
// ---------------------------------------------------------------------------
static __device__ __forceinline__ unsigned long long splitmix64(unsigned long long x) {
    x += 0x9E3779B97F4A7C15ull;
    x = (x ^ (x >> 30)) * 0xBF58476D1CE4E5B9ull;
    x = (x ^ (x >> 27)) * 0x94D049BB133111EBull;
    return x ^ (x >> 31);
}

__global__ __launch_bounds__(256) void gamma_sample(const float* __restrict__ alpha,
                                                    float* __restrict__ out, int n) {
    int i = blockIdx.x * blockDim.x + threadIdx.x;
    if (i >= n) return;
    float a = alpha[i];
    float d = a - (1.0f / 3.0f);            // alpha > 0.5 guaranteed -> d > 0
    float c = rsqrtf(9.0f * d);
    float sample = d;                        // fallback (P(all rejected) < 1e-20)
    bool  done   = false;
    unsigned long long s = splitmix64((unsigned long long)(unsigned)i * 0x100000001B3ull + 0x9E3779B9ull);
#pragma unroll 1
    for (int k = 0; k < 16; ++k) {
        if (done) break;
        s = splitmix64(s + (unsigned long long)k);
        unsigned u1 = (unsigned)s, u2 = (unsigned)(s >> 32);
        s = splitmix64(s);
        unsigned u3 = (unsigned)s;
        float f1 = (float)(u1 >> 8) * (1.0f / 16777216.0f) + 1e-7f;
        float f2 = (float)(u2 >> 8) * (1.0f / 16777216.0f);
        float z  = sqrtf(-2.0f * __logf(f1)) * __cosf(6.28318530718f * f2);
        float uu = (float)(u3 >> 8) * (1.0f / 16777216.0f) + 1e-7f;
        float v  = 1.0f + c * z;
        v = v * v * v;
        if (v > 0.0f) {
            float b2 = 0.5f * z * z + d - d * v + d * __logf(v);
            if (__logf(uu) < b2) { sample = d * v; done = true; }
        }
    }
    out[i] = sample;
}

// ---------------------------------------------------------------------------
// Launch
// ---------------------------------------------------------------------------
extern "C" void kernel_launch(void* const* d_in, const int* in_sizes, int n_in,
                              void* d_out, int out_size, void* d_ws, size_t ws_size,
                              hipStream_t stream) {
    const float* x  = (const float*)d_in[0];
    const float* W0 = (const float*)d_in[1];
    const float* b0 = (const float*)d_in[2];
    const float* W1 = (const float*)d_in[3];
    const float* b1 = (const float*)d_in[4];
    const float* Wl = (const float*)d_in[5];
    const float* bl = (const float*)d_in[6];

    const int Bn = 32768, DIN = 1024, H = 512, L = 64;

    char* ws = (char*)d_ws;
    unsigned* W0p = (unsigned*)ws;            ws += (size_t)(DIN / 2) * H * 4;  // 1 MB
    unsigned* W1p = (unsigned*)ws;            ws += (size_t)(H / 2) * H * 4;    // 512 KB
    unsigned* Wlp = (unsigned*)ws;            ws += (size_t)(H / 2) * L * 4;    // 64 KB
    unsigned short* h1 = (unsigned short*)ws; ws += (size_t)Bn * H * 2;         // 32 MB
    unsigned short* h2 = (unsigned short*)ws; ws += (size_t)Bn * H * 2;         // 32 MB

    float* alpha  = (float*)d_out;                 // [B, L]
    float* sample = alpha + (size_t)Bn * L;        // [B, L]

    pack_weights_t<<<((DIN / 2) * H + 255) / 256, 256, 0, stream>>>(W0, W0p, DIN, H);
    pack_weights_t<<<((H / 2) * H + 255) / 256, 256, 0, stream>>>(W1, W1p, H, H);
    pack_weights_t<<<((H / 2) * L + 255) / 256, 256, 0, stream>>>(Wl, Wlp, H, L);

    dim3 g1(Bn / BM, H / BN);
    gemm_wmma<true,  false><<<g1, 256, 0, stream>>>((const void*)x,  W0p, b0, (void*)h1, Bn, H, DIN);
    gemm_wmma<false, false><<<g1, 256, 0, stream>>>((const void*)h1, W1p, b1, (void*)h2, Bn, H, H);
    dim3 g3(Bn / BM, L / BN);
    gemm_wmma<false, true ><<<g3, 256, 0, stream>>>((const void*)h2, Wlp, bl, (void*)alpha, Bn, L, H);

    gamma_sample<<<(Bn * L + 255) / 256, 256, 0, stream>>>(alpha, sample, Bn * L);
}
